// dmc_4827543241005
// MI455X (gfx1250) — matrix operand order repaired and checked by reading, not compiled, not run on hardware
//
#include <hip/hip_runtime.h>
#include <hip/hip_bf16.h>
#include <math.h>

// ---- sizes (from reference config) ----
#define NR 4096          // N_REACH
#define T_SEC 3600.0f
#define DISCHARGE_LB 1e-4f
#define SLOPE_LB 1e-4f
#define VEL_LB 0.3f
#define VEL_UB 15.0f

typedef __attribute__((ext_vector_type(16))) __bf16 v16bf;
typedef __attribute__((ext_vector_type(8)))  float v8f;

// 3-term bf16 residual split: h + m + l reconstructs f32 to ~full precision,
// and products with binary A are exact in the WMMA f32 accumulator.
__device__ __forceinline__ void bf16_split3(float v, __bf16& h, __bf16& m, __bf16& l) {
  h = (__bf16)v;
  float r1 = v - (float)h;
  m = (__bf16)r1;
  l = (__bf16)(r1 - (float)m);
}

// ---------------------------------------------------------------------------
// Kernel 1: per-reach Muskingum coefficients c1..c4, d0 = q_prime[0], out[0],
// plus the bf16 3-way split of d0 used by the WMMA GEMV.
// ---------------------------------------------------------------------------
__global__ void dmc_prep_kernel(const float* __restrict__ qp0,
                                const float* __restrict__ raw_n,
                                const float* __restrict__ raw_q,
                                const float* __restrict__ raw_p,
                                const float* __restrict__ width,
                                const float* __restrict__ length,
                                const float* __restrict__ slope,
                                const float* __restrict__ xst,
                                float* __restrict__ c1, float* __restrict__ c2,
                                float* __restrict__ c3, float* __restrict__ c4,
                                float* __restrict__ dvec,
                                __bf16* __restrict__ dh, __bf16* __restrict__ dm,
                                __bf16* __restrict__ dl, float* __restrict__ out) {
  int i = blockIdx.x * blockDim.x + threadIdx.x;
  if (i >= NR) return;
  float n    = 0.01f + raw_n[i] * (0.3f - 0.01f);
  float q_sp = 1.5f  + raw_q[i] * (3.0f - 1.5f);
  float p_sp = 0.5f  + raw_p[i] * (2.0f - 0.5f);
  float s0   = fmaxf(slope[i], SLOPE_LB);
  float depth = logf(width[i] / p_sp) / logf(q_sp);
  float v = (1.0f / n) * powf(depth, 0.6666667f) * sqrtf(s0);
  float c = fminf(fmaxf(v, VEL_LB), VEL_UB) * (5.0f / 3.0f);
  float k = length[i] / c;
  float x = xst[0];
  float denom = 2.0f * k * (1.0f - x) + T_SEC;
  c1[i] = (T_SEC - 2.0f * k * x) / denom;
  c2[i] = (T_SEC + 2.0f * k * x) / denom;
  c3[i] = (2.0f * k * (1.0f - x) - T_SEC) / denom;
  c4[i] = 2.0f * T_SEC / denom;
  float d0 = qp0[i];
  dvec[i] = d0;                       // unclamped, per reference
  __bf16 h, m, l;
  bf16_split3(d0, h, m, l);
  dh[i] = h; dm[i] = m; dl[i] = l;
  if (i == NR - 1) out[0] = fmaxf(d0, DISCHARGE_LB);
}

// ---------------------------------------------------------------------------
// Kernel 2: b = c2*(A@d) + c3*d + c4*max(q_lat,lb) via bf16 WMMA.
// One wave (32 lanes) per 16-row tile.  B columns 0..2 hold the precomputed
// hi/mid/lo bf16 split of d; inner loop is pure loads + v_wmma.
// K only walks up to the tile diagonal (A strictly lower-triangular).
// ---------------------------------------------------------------------------
__global__ __launch_bounds__(32) void dmc_gemv_wmma_kernel(
    const float* __restrict__ adj, const float* __restrict__ d,
    const __bf16* __restrict__ dh, const __bf16* __restrict__ dm,
    const __bf16* __restrict__ dl, const float* __restrict__ qp,
    const float* __restrict__ c2, const float* __restrict__ c3,
    const float* __restrict__ c4, float* __restrict__ bout) {
  const int lane = threadIdx.x;          // 0..31, wave32
  const int m0   = blockIdx.x * 16;      // first row of tile
  const int mrow = m0 + (lane & 15);     // A-operand row for this lane
  const int ahalf = (lane & 16) ? 8 : 0; // A K-sub-block select
  const int bhalf = (lane & 16) ? 8 : 0; // B K-sub-block select
  const int ncol = lane & 15;            // B column owned by this lane
  const bool bact = (ncol < 3);
  const __bf16* bsrc = (ncol == 1) ? dm : (ncol == 2) ? dl : dh;

  v8f acc = {0.f, 0.f, 0.f, 0.f, 0.f, 0.f, 0.f, 0.f};
  v16bf bm;
#pragma unroll
  for (int e = 0; e < 16; ++e) bm[e] = (__bf16)0.0f;

  int kend = ((m0 + 16) + 31) & ~31;     // cover all j < m0+16, 32-aligned
  if (kend > NR) kend = NR;

  const float* arowbase = adj + (size_t)mrow * NR + ahalf;
  for (int k0 = 0; k0 < kend; k0 += 32) {
    // ---- A operand: 16x32 bf16, ISA wave32 layout ----
    const float* arow = arowbase + k0;
    if (k0 + 32 < kend) __builtin_prefetch(arow + 32, 0, 1);  // next panel -> L0
    v16bf a;
#pragma unroll
    for (int e = 0; e < 8; ++e) {
      a[e]     = (__bf16)arow[e];
      a[e + 8] = (__bf16)arow[16 + e];
    }
    // ---- B operand: 32x16 bf16; col0=hi(d), col1=mid, col2=lo ----
    if (bact) {
      const __bf16* bk = bsrc + k0 + bhalf;
#pragma unroll
      for (int e = 0; e < 16; ++e) bm[e] = bk[(e < 8) ? e : e + 8];
    }
    acc = __builtin_amdgcn_wmma_f32_16x16x32_bf16(
        false, a, false, bm, (short)0, acc, false, false);
  }

  // ---- recombine D columns 0..2 per row; apply Muskingum combination ----
  __shared__ float red[32][8];
#pragma unroll
  for (int r = 0; r < 8; ++r) red[lane][r] = acc[r];
  __syncthreads();
  if (lane < 16) {
    int M = lane;
    int hl = (M < 8) ? 0 : 16;   // lane holding N=0 for this M group
    int r  = M & 7;              // C/D VGPR index
    float y = red[hl][r] + red[hl + 1][r] + red[hl + 2][r];  // (A@d)[m0+M]
    int i = m0 + M;
    float ql = fmaxf(qp[i], DISCHARGE_LB);
    bout[i] = c2[i] * y + c3[i] * d[i] + c4[i] * ql;
  }
}

// ---------------------------------------------------------------------------
// Kernel 3: forward substitution x = b + diag(c1) A x  (unit lower tri),
// single workgroup.  x/S/b/c1 live in LDS.  32-wide diagonal blocks are
// solved serially by wave 0 (wave32 shfl reduce); the rank-32 panel update
// of all future rows is done with bf16 WMMA by all 32 waves (one
// v_wmma_f32_16x16x32_bf16 per 16-row tile, B = bf16 split of x_blk in LDS).
// Finalize writes clamped d, its bf16 split, and out[t].
// ---------------------------------------------------------------------------
__global__ __launch_bounds__(1024) void dmc_solve_kernel(
    const float* __restrict__ adj, const float* __restrict__ bvec,
    const float* __restrict__ c1g, float* __restrict__ dvec,
    __bf16* __restrict__ dh, __bf16* __restrict__ dm, __bf16* __restrict__ dl,
    float* __restrict__ out, int t) {
  __shared__ float xsh[NR];
  __shared__ float Ssh[NR];
  __shared__ float bb[NR];
  __shared__ float c1s[NR];
  __shared__ float red[32][32][8];   // per-wave D recombine scratch (32 KB)
  __shared__ __bf16 xh[32], xm[32], xl[32];
  const int tid  = threadIdx.x;
  const int w    = tid >> 5;
  const int lane = tid & 31;
  const int ahalf = (lane & 16) ? 8 : 0;
  const int ncol  = lane & 15;
  const int bhalf = (lane & 16) ? 8 : 0;

  for (int i = tid; i < NR; i += 1024) {
    Ssh[i] = 0.0f;
    bb[i] = bvec[i];
    c1s[i] = c1g[i];
  }
  __syncthreads();

  volatile float* xv = xsh;
  for (int jb = 0; jb < NR / 32; ++jb) {
    const int base = jb << 5;
    if (tid < 32) {
      // ---- solve rows base..base+31 sequentially (wave32 coop dot) ----
      for (int r = 0; r < 32; ++r) {
        int i = base + r;
        float p = 0.0f;
        if (tid < r) p = adj[(size_t)i * NR + base + tid] * xv[base + tid];
#pragma unroll
        for (int off = 16; off; off >>= 1) p += __shfl_xor(p, off, 32);
        if (tid == 0) xv[i] = bb[i] + c1s[i] * (Ssh[i] + p);
      }
      // ---- bf16 3-way split of x_blk for the WMMA panel update ----
      __bf16 h, m, l;
      bf16_split3(xv[base + tid], h, m, l);
      xh[tid] = h; xm[tid] = m; xl[tid] = l;
    }
    __syncthreads();

    // ---- WMMA panel update: S[i] += A[i, base..base+31] . x_blk ----
    // B operand (32x16 bf16) is identical for every tile in this block.
    v16bf bm;
#pragma unroll
    for (int e = 0; e < 16; ++e) bm[e] = (__bf16)0.0f;
    if (ncol < 3) {
      const __bf16* src = (ncol == 1) ? xm : (ncol == 2) ? xl : xh;
#pragma unroll
      for (int e = 0; e < 16; ++e) bm[e] = src[bhalf + ((e < 8) ? e : e + 8)];
    }
    for (int tile = 2 * jb + 2 + w; tile < NR / 16; tile += 32) {
      const int row = tile * 16 + (lane & 15);
      const float* arow = adj + (size_t)row * NR + base + ahalf;
      v16bf a;
#pragma unroll
      for (int e = 0; e < 8; ++e) {
        a[e]     = (__bf16)arow[e];
        a[e + 8] = (__bf16)arow[16 + e];
      }
      v8f acc = {0.f, 0.f, 0.f, 0.f, 0.f, 0.f, 0.f, 0.f};
      acc = __builtin_amdgcn_wmma_f32_16x16x32_bf16(
          false, a, false, bm, (short)0, acc, false, false);
#pragma unroll
      for (int r = 0; r < 8; ++r) red[w][lane][r] = acc[r];
      // same-wave LDS ops are in-order; lanes 0..15 recombine cols 0..2
      if (lane < 16) {
        int hl = (lane < 8) ? 0 : 16;
        int r  = lane & 7;
        float y = red[w][hl][r] + red[w][hl + 1][r] + red[w][hl + 2][r];
        Ssh[tile * 16 + lane] += y;   // unique writer per row
      }
    }
    __syncthreads();
  }

  // ---- finalize: clamp, write d + its bf16 split, emit gage value ----
  for (int i = tid; i < NR; i += 1024) {
    float dn = fmaxf(xsh[i], DISCHARGE_LB);
    dvec[i] = dn;
    __bf16 h, m, l;
    bf16_split3(dn, h, m, l);
    dh[i] = h; dm[i] = m; dl[i] = l;
  }
  if (tid == 0) out[t] = fmaxf(xsh[NR - 1], DISCHARGE_LB);
}

// ---------------------------------------------------------------------------
extern "C" void kernel_launch(void* const* d_in, const int* in_sizes, int n_in,
                              void* d_out, int out_size, void* d_ws, size_t ws_size,
                              hipStream_t stream) {
  const float* q_prime = (const float*)d_in[0];  // [720, 4096]
  const float* raw_n   = (const float*)d_in[1];
  const float* raw_q   = (const float*)d_in[2];
  const float* raw_p   = (const float*)d_in[3];
  const float* width   = (const float*)d_in[4];
  const float* length  = (const float*)d_in[5];
  const float* slope   = (const float*)d_in[6];
  const float* adj     = (const float*)d_in[7];  // [4096, 4096]
  const float* xst     = (const float*)d_in[8];  // scalar
  float* out = (float*)d_out;                    // 720 floats

  float* ws = (float*)d_ws;   // 6*4096 f32 + 3*4096 bf16 = 120 KB
  float* c1 = ws + 0 * NR;
  float* c2 = ws + 1 * NR;
  float* c3 = ws + 2 * NR;
  float* c4 = ws + 3 * NR;
  float* dv = ws + 4 * NR;
  float* bv = ws + 5 * NR;
  __bf16* dh = (__bf16*)(ws + 6 * NR);
  __bf16* dm = dh + NR;
  __bf16* dl = dm + NR;

  const int nT = in_sizes[0] / NR;  // 720

  dmc_prep_kernel<<<NR / 256, 256, 0, stream>>>(
      q_prime, raw_n, raw_q, raw_p, width, length, slope, xst,
      c1, c2, c3, c4, dv, dh, dm, dl, out);

  for (int t = 1; t < nT; ++t) {
    dmc_gemv_wmma_kernel<<<NR / 16, 32, 0, stream>>>(
        adj, dv, dh, dm, dl, q_prime + (size_t)(t - 1) * NR, c2, c3, c4, bv);
    dmc_solve_kernel<<<1, 1024, 0, stream>>>(adj, bv, c1, dv, dh, dm, dl, out, t);
  }
}
